// self_attention_cam_9844065042954
// MI455X (gfx1250) — matrix operand order repaired and checked by reading, not compiled, not run on hardware
//
#include <hip/hip_runtime.h>

typedef unsigned int u32;
typedef unsigned short u16;
typedef __attribute__((ext_vector_type(16))) __bf16 v16bf;
typedef __attribute__((ext_vector_type(8)))  float  v8f;
typedef __attribute__((ext_vector_type(8)))  u32    v8u;

#define NB     8
#define CIN    256
#define NPIX   4096
#define ADIM   128
#define NCLS   21
#define NTILE  256                 // NPIX / 16
#define L2E    1.44269504088896340736f

// ---------- workspace layout (in 4-byte units) ----------
#define WS_QB   0u                 // NB*NTILE*32*32 dwords = 2,097,152 (8 MB) bf16 Q, B-layout
#define WS_KA   2097152u           // 8 MB bf16 K, A-layout
#define WS_VA   4194304u           // NB*2*128*32*8 = 524,288 dwords (2 MB) bf16 V, A-layout
#define WS_ATT  4718592u           // NB*21*4096 floats = 688,128 (att+residual, then y in-place)
#define WS_STAT 5406720u           // 64 floats: sum[32], sumsq[32]
#define WS_SC   (WS_STAT + 64u)    // 64 floats: scale[32], shift[32]

__device__ inline u16 f2bf(float f) {
    u32 u = __builtin_bit_cast(u32, f);
    u += 0x7FFFu + ((u >> 16) & 1u);        // round to nearest even
    return (u16)(u >> 16);
}
__device__ inline u32 pk_bf16(float lo, float hi) {
    return (u32)f2bf(lo) | ((u32)f2bf(hi) << 16);
}
__device__ inline v16bf ld16(const u32* __restrict__ p) {
    uint4 a = *(const uint4*)p;
    uint4 b = *(const uint4*)(p + 4);
    v8u u;
    u[0]=a.x; u[1]=a.y; u[2]=a.z; u[3]=a.w;
    u[4]=b.x; u[5]=b.y; u[6]=b.z; u[7]=b.w;
    return __builtin_bit_cast(v16bf, u);
}
// Cross-half (xor 16) exchange as a pure VALU op: v_permlanex16_b32 with
// identity lane select. Avoids ds_bpermute + s_wait_dscnt on the hot path.
__device__ inline float xhalf(float x) {
    int i = __builtin_bit_cast(int, x);
    int r = __builtin_amdgcn_permlanex16(i, i, 0x76543210, (int)0xfedcba98u, false, false);
    return __builtin_bit_cast(float, r);
}

// ---------------------------------------------------------------------------
// Phase 1a: q = Wq x + bq (scaled by 1/sqrt(128)), k = Wk x + bk; bf16,
// pre-swizzled into WMMA B-layout (Q) and A-layout (K).
// grid: NB * (NPIX/64) = 512 blocks, 256 threads.
// ---------------------------------------------------------------------------
__global__ __launch_bounds__(256) void proj_qk_kernel(
    const float* __restrict__ x, const float* __restrict__ Wq,
    const float* __restrict__ bq, const float* __restrict__ Wk,
    const float* __restrict__ bk, u32* __restrict__ qb, u32* __restrict__ ka)
{
    __shared__ float xs[CIN * 64];
    const int tid = threadIdx.x;
    const int blk = blockIdx.x;
    const int b  = blk >> 6;
    const int n0 = (blk & 63) * 64;
    const float* xb = x + (size_t)b * CIN * NPIX + n0;
    // stage X tile with b128 loads: 4096 float4, 256 threads x 16 iters
    #pragma unroll 4
    for (int i = 0; i < 16; ++i) {
        int idx4 = i * 256 + tid;
        int c = idx4 >> 4, p4 = idx4 & 15;
        float4 v = *(const float4*)(xb + c * NPIX + p4 * 4);
        *(float4*)(xs + c * 64 + p4 * 4) = v;
    }
    __syncthreads();
    for (int j = 0; j < 64; ++j) {
        int oid = j * 256 + tid;
        int dout = oid >> 6, p = oid & 63;
        bool isq = dout < ADIM;
        int d = isq ? dout : dout - ADIM;
        const float4* w4 = (const float4*)((isq ? Wq : Wk) + d * CIN);
        float acc = isq ? bq[d] : bk[d];
        #pragma unroll 4
        for (int c4 = 0; c4 < CIN / 4; ++c4) {
            float4 w = w4[c4];
            const float* xc = xs + (c4 * 4) * 64 + p;
            acc += w.x * xc[0] + w.y * xc[64] + w.z * xc[128] + w.w * xc[192];
        }
        if (isq) acc *= 0.08838834764831845f;   // 1/sqrt(128) folded into q
        const int n = n0 + p;
        const int t  = n >> 4, m = n & 15;
        const int sd = d >> 5, dd = d & 31, hi = dd & 1;
        u32 dwi;
        u32* dst;
        if (isq) {  // B-layout 32x16: lane = n%16 + 16*(dd/16), vgpr = (dd%16)/2
            int lane = m + ((dd >> 4) << 4);
            int V = (sd << 3) + ((dd & 15) >> 1);
            dwi = ((((u32)b * NTILE + t) * 32 + lane) << 5) + V;
            dst = qb;
        } else {    // A-layout 16x32: lane = key%16 + 16*(dd/16), vgpr = (dd%16)/2
            int lane = m + ((dd >> 4) << 4);
            int V = (sd << 3) + ((dd & 15) >> 1);
            dwi = ((((u32)b * NTILE + t) * 32 + lane) << 5) + V;
            dst = ka;
        }
        reinterpret_cast<u16*>(dst)[dwi * 2 + hi] = f2bf(acc);
    }
}

// ---------------------------------------------------------------------------
// Phase 1b: V (logits, 21 channels padded to 32) into WMMA A-layout bf16.
// One thread per output dword. grid: 2048 x 256.
// ---------------------------------------------------------------------------
__global__ __launch_bounds__(256) void build_va_kernel(
    const float* __restrict__ logits, u32* __restrict__ va)
{
    int idx = blockIdx.x * 256 + threadIdx.x;     // (((b*2+ct)*128+s)*32+lane)*8+v
    if (idx >= NB * 2 * 128 * 32 * 8) return;
    int v    = idx & 7;
    int lane = (idx >> 3) & 31;
    int s    = (idx >> 8) & 127;
    int ct   = (idx >> 15) & 1;
    int b    = idx >> 16;
    int M = lane & 15, h = lane >> 4;
    int c = ct * 16 + M;
    int r0 = ((v & 7) << 1) + (h << 4);
    int k0 = (s << 5) + r0;
    float lo = 0.f, hi = 0.f;
    if (c < NCLS) {
        const float* p = logits + (((size_t)b * NCLS + c) << 12);
        lo = p[k0]; hi = p[k0 + 1];
    }
    va[idx] = pk_bf16(lo, hi);
}

// ---------------------------------------------------------------------------
// Phase 2: streaming (flash) attention + PV, all matrix work in WMMA bf16.
// One wave (32 threads) per (batch, 16-query tile). grid = 8*256 = 2048.
// ---------------------------------------------------------------------------
__global__ __launch_bounds__(32) void attn_kernel(
    const u32* __restrict__ qb, const u32* __restrict__ ka,
    const u32* __restrict__ va, const float* __restrict__ logits,
    float* __restrict__ att)
{
    const int lane = threadIdx.x;
    const int bq_  = blockIdx.x;
    const int b  = bq_ >> 8;
    const int qt = bq_ & 255;
    const int nn = lane & 15;
    const int h  = lane >> 4;
    const bool hihalf = (lane & 16) != 0;

    // resident Q tile (B-layout), 32 dwords per lane
    const u32* qp = qb + ((((u32)b * NTILE + qt) * 32 + lane) << 5);
    v16bf qv[4];
    #pragma unroll
    for (int sd = 0; sd < 4; ++sd) qv[sd] = ld16(qp + sd * 8);

    v8f o0 = {}, o1 = {};
    float mrun = -3.0e38f, lrun = 0.f;

    const u32* kbase = ka + ((((u32)b * NTILE) * 32 + lane) << 5);
    const u32* v0base = va + ((((u32)(b * 2 + 0) * 128) * 32 + lane) << 3);
    const u32* v1base = va + ((((u32)(b * 2 + 1) * 128) * 32 + lane) << 3);

    for (int step = 0; step < 128; ++step) {
        const u32* k0p = kbase + (u32)step * 2048;   // 2 tiles * 32 lanes * 32 dwords
        const u32* k1p = k0p + 1024;
        __builtin_prefetch(k0p + 2048, 0, 3);        // next step (L2 hint)

        v8f s0 = {}, s1 = {};
        #pragma unroll
        for (int sd = 0; sd < 4; ++sd)
            s0 = __builtin_amdgcn_wmma_f32_16x16x32_bf16(
                false, ld16(k0p + sd * 8), false, qv[sd], (short)0, s0, false, false);
        #pragma unroll
        for (int sd = 0; sd < 4; ++sd)
            s1 = __builtin_amdgcn_wmma_f32_16x16x32_bf16(
                false, ld16(k1p + sd * 8), false, qv[sd], (short)0, s1, false, false);

        // ---- online softmax (per query = per lane column) ----
        float mt = -3.0e38f;
        #pragma unroll
        for (int r = 0; r < 8; ++r) mt = fmaxf(mt, fmaxf(s0[r], s1[r]));
        mt = fmaxf(mt, xhalf(mt));
        float mnew  = fmaxf(mrun, mt);
        float alpha = __builtin_amdgcn_exp2f((mrun - mnew) * L2E);
        float p0[8], p1[8], ts = 0.f;
        #pragma unroll
        for (int r = 0; r < 8; ++r) {
            p0[r] = __builtin_amdgcn_exp2f((s0[r] - mnew) * L2E);
            p1[r] = __builtin_amdgcn_exp2f((s1[r] - mnew) * L2E);
            ts += p0[r] + p1[r];
        }
        ts += xhalf(ts);
        lrun = lrun * alpha + ts;
        mrun = mnew;
        #pragma unroll
        for (int r = 0; r < 8; ++r) { o0[r] *= alpha; o1[r] *= alpha; }

        // ---- repack P (two 16x16 D-layout tiles) into one 32x16 B-layout ----
        u32 pb[8];
        #pragma unroll
        for (int v = 0; v < 4; ++v) {
            float xa = xhalf(p1[2 * v]);
            float xb = xhalf(p1[2 * v + 1]);
            pb[v] = pk_bf16(hihalf ? xa : p0[2 * v], hihalf ? xb : p0[2 * v + 1]);
        }
        #pragma unroll
        for (int v = 0; v < 4; ++v) {
            float xa = xhalf(p0[2 * v]);
            float xb = xhalf(p0[2 * v + 1]);
            pb[v + 4] = pk_bf16(hihalf ? p1[2 * v] : xa, hihalf ? p1[2 * v + 1] : xb);
        }
        v8u pu;
        #pragma unroll
        for (int v = 0; v < 8; ++v) pu[v] = pb[v];
        v16bf pbv = __builtin_bit_cast(v16bf, pu);

        // ---- O += V_tile x P (channels 0..15 and 16..31) ----
        o0 = __builtin_amdgcn_wmma_f32_16x16x32_bf16(
            false, ld16(v0base + (u32)step * 256), false, pbv, (short)0, o0, false, false);
        o1 = __builtin_amdgcn_wmma_f32_16x16x32_bf16(
            false, ld16(v1base + (u32)step * 256), false, pbv, (short)0, o1, false, false);
    }

    const float invl = __builtin_amdgcn_rcpf(lrun);
    const int n = qt * 16 + nn;
    #pragma unroll
    for (int r = 0; r < 8; ++r) {
        int c0 = r + 8 * h;                                   // < 16
        size_t i0 = (((size_t)b * NCLS + c0) << 12) + n;
        att[i0] = o0[r] * invl + logits[i0];
        int c1 = 16 + c0;
        if (c1 < NCLS) {
            size_t i1 = (((size_t)b * NCLS + c1) << 12) + n;
            att[i1] = o1[r] * invl + logits[i1];
        }
    }
}

// ---------------------------------------------------------------------------
// Phase 3: Wo conv (21x21) + BN statistics + normalize.
// ---------------------------------------------------------------------------
__global__ void zero_stats_kernel(float* __restrict__ st)
{
    if (threadIdx.x < 64) st[threadIdx.x] = 0.f;
}

__global__ __launch_bounds__(256) void conv_bn_partial_kernel(
    float* __restrict__ y, const float* __restrict__ Wo,
    const float* __restrict__ bo, float* __restrict__ stats)
{
    __shared__ float ssum[NCLS], ssq[NCLS];
    const int tid = threadIdx.x;
    if (tid < NCLS) { ssum[tid] = 0.f; ssq[tid] = 0.f; }
    __syncthreads();
    const int pix = blockIdx.x * 256 + tid;     // 0..32767
    const int b = pix >> 12, n = pix & 4095;
    float a[NCLS], yv[NCLS];
    #pragma unroll
    for (int c = 0; c < NCLS; ++c) a[c] = y[(((size_t)b * NCLS + c) << 12) + n];
    #pragma unroll
    for (int c = 0; c < NCLS; ++c) {
        float acc = bo[c];
        #pragma unroll
        for (int j = 0; j < NCLS; ++j) acc += Wo[c * NCLS + j] * a[j];
        yv[c] = acc;
    }
    #pragma unroll
    for (int c = 0; c < NCLS; ++c) {
        y[(((size_t)b * NCLS + c) << 12) + n] = yv[c];   // in-place
        atomicAdd(&ssum[c], yv[c]);
        atomicAdd(&ssq[c], yv[c] * yv[c]);
    }
    __syncthreads();
    if (tid < NCLS) {
        atomicAdd(&stats[tid], ssum[tid]);
        atomicAdd(&stats[32 + tid], ssq[tid]);
    }
}

__global__ void bn_finalize_kernel(const float* __restrict__ stats,
                                   const float* __restrict__ gamma,
                                   const float* __restrict__ beta,
                                   float* __restrict__ sc)
{
    int c = threadIdx.x;
    if (c < NCLS) {
        const float invN = 1.0f / (float)(NB * NPIX);
        float mean = stats[c] * invN;
        float var  = stats[32 + c] * invN - mean * mean;
        float inv  = __builtin_amdgcn_rsqf(var + 1e-5f);
        float s    = gamma[c] * inv;
        sc[c]      = s;
        sc[32 + c] = beta[c] - mean * s;
    }
}

__global__ __launch_bounds__(256) void bn_apply_kernel(
    const float* __restrict__ y, const float* __restrict__ sc,
    float* __restrict__ out)
{
    int i = blockIdx.x * 256 + threadIdx.x;
    if (i >= NB * NCLS * NPIX) return;
    int c = (i >> 12) % NCLS;
    out[i] = y[i] * sc[c] + sc[32 + c];
}

// ---------------------------------------------------------------------------
extern "C" void kernel_launch(void* const* d_in, const int* in_sizes, int n_in,
                              void* d_out, int out_size, void* d_ws, size_t ws_size,
                              hipStream_t stream)
{
    const float* logits = (const float*)d_in[0];
    const float* hyper  = (const float*)d_in[1];
    const float* Wk     = (const float*)d_in[2];
    const float* bk     = (const float*)d_in[3];
    const float* Wq     = (const float*)d_in[4];
    const float* bq     = (const float*)d_in[5];
    const float* Wo     = (const float*)d_in[6];
    const float* bo     = (const float*)d_in[7];
    const float* gamma  = (const float*)d_in[8];
    const float* beta   = (const float*)d_in[9];
    float* out = (float*)d_out;

    u32*   wsu  = (u32*)d_ws;
    float* wsf  = (float*)d_ws;
    u32*   QB   = wsu + WS_QB;
    u32*   KA   = wsu + WS_KA;
    u32*   VA   = wsu + WS_VA;
    float* ATT  = wsf + WS_ATT;
    float* STAT = wsf + WS_STAT;
    float* SC   = wsf + WS_SC;

    proj_qk_kernel<<<NB * (NPIX / 64), 256, 0, stream>>>(hyper, Wq, bq, Wk, bk, QB, KA);
    build_va_kernel<<<2048, 256, 0, stream>>>(logits, VA);
    attn_kernel<<<NB * NTILE, 32, 0, stream>>>(QB, KA, VA, logits, ATT);
    zero_stats_kernel<<<1, 64, 0, stream>>>(STAT);
    conv_bn_partial_kernel<<<(NB * NPIX) / 256, 256, 0, stream>>>(ATT, Wo, bo, STAT);
    bn_finalize_kernel<<<1, 32, 0, stream>>>(STAT, gamma, beta, SC);
    bn_apply_kernel<<<(NB * NCLS * NPIX + 255) / 256, 256, 0, stream>>>(ATT, SC, out);
}
